// CMMBMambaSeg_35003983463222
// MI455X (gfx1250) — matrix operand order repaired and checked by reading, not compiled, not run on hardware
//
#include <hip/hip_runtime.h>
#include <hip/hip_bf16.h>
#include <math.h>

typedef __attribute__((ext_vector_type(16))) __bf16 v16bf;
typedef __attribute__((ext_vector_type(8)))  float  v8f;
typedef __attribute__((ext_vector_type(4)))  unsigned int v4u;
typedef __attribute__((ext_vector_type(8)))  int v8i;
typedef __attribute__((ext_vector_type(4)))  int v4i;

#if __has_builtin(__builtin_amdgcn_tensor_load_to_lds) && \
    __has_builtin(__builtin_amdgcn_s_wait_tensorcnt)
#define HAS_TDM 1
#else
#define HAS_TDM 0
#endif

static inline int cdiv_h(int a, int b) { return (a + b - 1) / b; }

// ---------------------------------------------------------------------------
// Stem: 7x7x7 conv, stride 2, pad 3, cin=1 -> 48, 64^3 -> 32^3. Direct (K=1
// per tap makes WMMA pointless here).
// ---------------------------------------------------------------------------
__global__ __launch_bounds__(256)
void k_stem(const float* __restrict__ x, const float* __restrict__ w,
            const float* __restrict__ b, float* __restrict__ y) {
  int idx = blockIdx.x * blockDim.x + threadIdx.x;          // 48 * 32768
  int l = idx & 32767, co = idx >> 15;
  int ox = l & 31, oy = (l >> 5) & 31, oz = l >> 10;
  const float* wc = w + co * 343;
  float acc = b[co];
  for (int kd = 0; kd < 7; ++kd) {
    int iz = oz * 2 - 3 + kd; if ((unsigned)iz >= 64u) continue;
    for (int kh = 0; kh < 7; ++kh) {
      int iy = oy * 2 - 3 + kh; if ((unsigned)iy >= 64u) continue;
      for (int kw = 0; kw < 7; ++kw) {
        int ix = ox * 2 - 3 + kw; if ((unsigned)ix >= 64u) continue;
        acc += wc[(kd * 7 + kh) * 7 + kw] * x[(iz * 64 + iy) * 64 + ix];
      }
    }
  }
  y[idx] = acc;
}

#if HAS_TDM
// Issue one TDM 2-D tile load: tile0 x tile1 elements of f32, row stride
// `stride_e` elements, into LDS at byte offset `lds_byte`. D# per ISA 8.3-8.5.
static __device__ __forceinline__ void tdm_load_2d(unsigned lds_byte,
                                                   const float* gaddr,
                                                   int td0, int td1,
                                                   int tile0, int tile1,
                                                   int stride_e) {
  if (td0 < 0) td0 = 0;
  if (td1 < 0) td1 = 0;
  unsigned long long ga = (unsigned long long)(size_t)gaddr;
  unsigned long long st = (unsigned long long)(unsigned)stride_e;
  v4u g0;
  g0[0] = 1u;                                              // count=1, user mode
  g0[1] = lds_byte;                                        // lds_addr (bytes)
  g0[2] = (unsigned)(ga & 0xffffffffull);                  // global_addr lo
  g0[3] = (unsigned)((ga >> 32) & 0x1ffffffull) | (2u << 30);  // hi | type=2
  v8i g1;
  g1[0] = (int)(2u << 16);                                 // data_size = 4B
  g1[1] = (int)(((unsigned)td0 & 0xffffu) << 16);          // tensor_dim0[15:0]
  g1[2] = (int)((((unsigned)td0 >> 16) & 0xffffu) |
                (((unsigned)td1 & 0xffffu) << 16));        // td0 hi | td1 lo
  g1[3] = (int)((((unsigned)td1 >> 16) & 0xffffu) |
                (((unsigned)tile0 & 0xffffu) << 16));      // td1 hi | tile_dim0
  g1[4] = (int)((unsigned)tile1 & 0xffffu);                // tile_dim1 (dim2=0)
  g1[5] = (int)(unsigned)(st & 0xffffffffull);             // dim0_stride lo
  g1[6] = (int)(unsigned)((st >> 32) & 0xffffull);         // dim0_stride hi
  g1[7] = 0;
  v4i gz = {0, 0, 0, 0};
#if defined(__clang_major__) && __clang_major__ >= 23
  v8i gz8 = {0, 0, 0, 0, 0, 0, 0, 0};
  __builtin_amdgcn_tensor_load_to_lds(g0, g1, gz, gz, gz8, 0);
#else
  __builtin_amdgcn_tensor_load_to_lds(g0, g1, gz, gz, 0);
#endif
}
#endif

// ---------------------------------------------------------------------------
// Implicit-GEMM conv via V_WMMA_F32_16X16X32_BF16 (wave32, CDNA5).
// out[co, l] = bias[co] + sum_{tap, ci} W[co,ci,tap] * in[ci, shift(l,tap)]
// Activations channel-major (C, L), L = s^3 (power of two, lw = log2 s).
// One wave owns a 16(co) x 32(voxel) f32 tile (two WMMAs sharing A).
// For 1x1 taps the B tile (32 ci x 32 voxels, row stride L) is DMA'd into
// LDS by the Tensor Data Mover, double-buffered (s_wait_tensorcnt 1/0).
// For spatial taps, fragments are gathered straight from global.
// ---------------------------------------------------------------------------
__global__ __launch_bounds__(128)
void k_conv_wmma(const float* __restrict__ in, const float* __restrict__ wgt,
                 const float* __restrict__ bias, float* __restrict__ out,
                 int cin, int cout, int ks, int stride, int pad,
                 int inD, int inH, int inW, int outD, int outH, int outW,
                 int lw) {
#if HAS_TDM
  __shared__ float Bs[4][2][32 * 32];     // [wave][buffer][ci][voxel]
#endif
  const int lane = threadIdx.x & 31;
  const int wv   = threadIdx.x >> 5;
  const int Lout = outD * outH * outW;
  const int mbase = blockIdx.y * 16;
  const int nbase = blockIdx.x * 128 + wv * 32;
  const int inHW = inH * inW;
  const size_t inDHW = (size_t)inD * inHW;
  const int ks3 = ks * ks * ks;
  const int smask = outW - 1;
  const int nB = lane & 15;
  const int kHalf = lane >> 4;            // which K-half this lane holds

  // Decode this lane's two output columns (shift/mask: dims are 2^lw).
  int lB[2], od[2], oh[2], ow[2];
  bool okl[2];
#pragma unroll
  for (int c = 0; c < 2; ++c) {
    lB[c] = nbase + 16 * c + nB;
    okl[c] = lB[c] < Lout;
    ow[c] = lB[c] & smask;
    oh[c] = (lB[c] >> lw) & smask;
    od[c] = lB[c] >> (2 * lw);
  }

  v8f acc0 = {}, acc1 = {};
  const int nchunk = (cin + 31) >> 5;

#if HAS_TDM
  if (ks3 == 1 && stride == 1) {   // pure GEMM: TDM-staged, double-buffered
    unsigned lds0 = (unsigned)(size_t)(void*)&Bs[wv][0][0];
    unsigned lds1 = (unsigned)(size_t)(void*)&Bs[wv][1][0];
    tdm_load_2d(lds0, in + (size_t)0 * Lout + nbase, Lout - nbase, cin,
                32, 32, Lout);
    for (int ci = 0; ci < nchunk; ++ci) {
      const int cb = ci << 5;
      if (ci + 1 < nchunk) {
        tdm_load_2d((ci & 1) ? lds0 : lds1,
                    in + (size_t)(cb + 32) * Lout + nbase,
                    Lout - nbase, cin - (cb + 32), 32, 32, Lout);
        __builtin_amdgcn_s_wait_tensorcnt(1);   // current tile landed
      } else {
        __builtin_amdgcn_s_wait_tensorcnt(0);
      }
      // A fragment straight from global weights (cout,cin) row-major.
      v16bf a;
      {
        const int m = lane & 15, ka = kHalf * 8;
        const int co = mbase + m;
        const float* wrow = wgt + (size_t)co * cin + cb;
        const bool okm = co < cout;
#pragma unroll
        for (int j = 0; j < 8; ++j) {
          int c0 = ka + j, c1 = ka + 16 + j;
          a[j]     = (okm && cb + c0 < cin) ? (__bf16)wrow[c0] : (__bf16)0.f;
          a[8 + j] = (okm && cb + c1 < cin) ? (__bf16)wrow[c1] : (__bf16)0.f;
        }
      }
      const float* tile = &Bs[wv][ci & 1][0];
      v16bf b0, b1;
      {
        const int kb = kHalf * 8;
#pragma unroll
        for (int j = 0; j < 16; ++j) {
          const int kk = kb + (j & 7) + ((j >> 3) << 4);
          b0[j] = (__bf16)tile[kk * 32 + nB];
          b1[j] = (__bf16)tile[kk * 32 + 16 + nB];
        }
      }
      acc0 = __builtin_amdgcn_wmma_f32_16x16x32_bf16(false, a, false, b0,
                                                     (short)0, acc0, false, false);
      acc1 = __builtin_amdgcn_wmma_f32_16x16x32_bf16(false, a, false, b1,
                                                     (short)0, acc1, false, false);
    }
  } else
#endif
  {
    for (int t = 0; t < ks3; ++t) {
      const int kd = t / (ks * ks), kh = (t / ks) % ks, kw = t % ks;
      // Per-column shifted input position for this tap.
      int iofs[2];
      bool okp[2];
#pragma unroll
      for (int c = 0; c < 2; ++c) {
        int iz = od[c] * stride - pad + kd;
        int iy = oh[c] * stride - pad + kh;
        int ix = ow[c] * stride - pad + kw;
        okp[c] = okl[c] && iz >= 0 && iz < inD && iy >= 0 && iy < inH &&
                 ix >= 0 && ix < inW;
        iofs[c] = (iz * inH + iy) * inW + ix;
      }
      for (int cb = 0; cb < cin; cb += 32) {
        v16bf a;
        {
          const int m = lane & 15, ka = kHalf * 8;
          const int co = mbase + m;
          const float* wrow = wgt + ((size_t)co * cin + cb) * ks3 + t;
          const bool okm = co < cout;
#pragma unroll
          for (int j = 0; j < 8; ++j) {
            int c0 = ka + j, c1 = ka + 16 + j;
            a[j]     = (okm && cb + c0 < cin)
                           ? (__bf16)wrow[(size_t)c0 * ks3] : (__bf16)0.f;
            a[8 + j] = (okm && cb + c1 < cin)
                           ? (__bf16)wrow[(size_t)c1 * ks3] : (__bf16)0.f;
          }
        }
        const int kb = kHalf * 8;
        v16bf b0, b1;
#pragma unroll
        for (int j = 0; j < 16; ++j) {
          int ci0 = cb + kb + (j & 7) + ((j >> 3) << 4);
          float v0 = (okp[0] && ci0 < cin) ? in[(size_t)ci0 * inDHW + iofs[0]] : 0.f;
          float v1 = (okp[1] && ci0 < cin) ? in[(size_t)ci0 * inDHW + iofs[1]] : 0.f;
          b0[j] = (__bf16)v0;
          b1[j] = (__bf16)v1;
        }
        acc0 = __builtin_amdgcn_wmma_f32_16x16x32_bf16(false, a, false, b0,
                                                       (short)0, acc0, false, false);
        acc1 = __builtin_amdgcn_wmma_f32_16x16x32_bf16(false, a, false, b1,
                                                       (short)0, acc1, false, false);
      }
    }
  }

  // Epilogue: C/D layout — lane n = lane&15, VGPR v holds row v + 8*kHalf.
#pragma unroll
  for (int v = 0; v < 8; ++v) {
    int m = mbase + v + kHalf * 8;
    if (m < cout) {
      float bv = bias ? bias[m] : 0.f;
      if (okl[0]) out[(size_t)m * Lout + lB[0]] = acc0[v] + bv;
      if (okl[1]) out[(size_t)m * Lout + lB[1]] = acc1[v] + bv;
    }
  }
}

// ---------------------------------------------------------------------------
// InstanceNorm3d (biased var), per channel over spatial L.
// ---------------------------------------------------------------------------
__global__ __launch_bounds__(256)
void k_inorm_stats(const float* __restrict__ x, float* __restrict__ mean,
                   float* __restrict__ rstd, int L) {
  const int c = blockIdx.x;
  __shared__ float s1[256], s2[256];
  float a = 0.f, q = 0.f;
  for (int i = threadIdx.x; i < L; i += 256) {
    float v = x[(size_t)c * L + i]; a += v; q += v * v;
  }
  s1[threadIdx.x] = a; s2[threadIdx.x] = q; __syncthreads();
  for (int s = 128; s > 0; s >>= 1) {
    if (threadIdx.x < s) { s1[threadIdx.x] += s1[threadIdx.x + s];
                           s2[threadIdx.x] += s2[threadIdx.x + s]; }
    __syncthreads();
  }
  if (threadIdx.x == 0) {
    float m = s1[0] / L, var = s2[0] / L - m * m;
    mean[c] = m; rstd[c] = rsqrtf(var + 1e-5f);
  }
}

__global__ __launch_bounds__(256)
void k_inorm_apply(const float* __restrict__ x, const float* __restrict__ mean,
                   const float* __restrict__ rstd, float* __restrict__ out,
                   int L, int relu) {
  const int c = blockIdx.y;
  const int i = blockIdx.x * 256 + threadIdx.x;
  if (i < L) {
    float v = (x[(size_t)c * L + i] - mean[c]) * rstd[c];
    if (relu) v = fmaxf(v, 0.f);
    out[(size_t)c * L + i] = v;
  }
}

// ---------------------------------------------------------------------------
// Elementwise helpers.
// ---------------------------------------------------------------------------
__global__ __launch_bounds__(256)
void k_add(const float* __restrict__ a, const float* __restrict__ b,
           float* __restrict__ o, int n) {
  int i = blockIdx.x * 256 + threadIdx.x;
  if (i < n) o[i] = a[i] + b[i];
}

__global__ __launch_bounds__(256)
void k_gelu(float* __restrict__ x, int n) {   // exact GELU
  int i = blockIdx.x * 256 + threadIdx.x;
  if (i < n) { float v = x[i]; x[i] = 0.5f * v * (1.f + erff(v * 0.70710678118654752f)); }
}

__global__ __launch_bounds__(256)
void k_softplus(float* __restrict__ x, int n) {
  int i = blockIdx.x * 256 + threadIdx.x;
  if (i < n) { float v = x[i]; x[i] = (v > 20.f) ? v : log1pf(expf(v)); }
}

// ---------------------------------------------------------------------------
// LayerNorm over channel dim, activations channel-major (C, L).
// ---------------------------------------------------------------------------
__global__ __launch_bounds__(128)
void k_lnorm(const float* __restrict__ x, const float* __restrict__ g,
             const float* __restrict__ b, float* __restrict__ out,
             int C, int L) {
  const int lane = threadIdx.x & 31;
  const int l = blockIdx.x * 4 + (threadIdx.x >> 5);
  if (l >= L) return;
  float a = 0.f, q = 0.f;
  for (int c = lane; c < C; c += 32) {
    float v = x[(size_t)c * L + l]; a += v; q += v * v;
  }
#pragma unroll
  for (int m = 16; m; m >>= 1) { a += __shfl_xor(a, m, 32); q += __shfl_xor(q, m, 32); }
  float mu = a / C, rs = rsqrtf(q / C - mu * mu + 1e-5f);
  for (int c = lane; c < C; c += 32) {
    float v = x[(size_t)c * L + l];
    out[(size_t)c * L + l] = (v - mu) * rs * g[c] + b[c];
  }
}

// ---------------------------------------------------------------------------
// Causal depthwise conv (D_CONV=4, left pad 3) + bias + SiLU. (C, L) layout.
// ---------------------------------------------------------------------------
__global__ __launch_bounds__(256)
void k_dwconv(const float* __restrict__ x, const float* __restrict__ w,
              const float* __restrict__ b, float* __restrict__ out,
              int C, int L) {
  long i = (long)blockIdx.x * 256 + threadIdx.x;
  if (i >= (long)C * L) return;
  int c = (int)(i / L), l = (int)(i % L);
  const float* wc = w + c * 4;
  float acc = b[c];
#pragma unroll
  for (int k = 0; k < 4; ++k) {
    int ll = l - 3 + k;
    if (ll >= 0) acc += wc[k] * x[(size_t)c * L + ll];
  }
  out[i] = acc / (1.f + expf(-acc));   // SiLU
}

// ---------------------------------------------------------------------------
// Selective scan. Wave32: lanes 0..15 = 16 states of channel 2*blk, lanes
// 16..31 = channel 2*blk+1. Sequential over L; cross-state reduction via
// shfl_xor within each 16-lane half. Fuses +u*D and *silu(z).
// ---------------------------------------------------------------------------
__global__ __launch_bounds__(32)
void k_scan(const float* __restrict__ dt, const float* __restrict__ Bm,
            const float* __restrict__ Cm, const float* __restrict__ u,
            const float* __restrict__ A_log, const float* __restrict__ Dv,
            const float* __restrict__ z, float* __restrict__ y, int L) {
  const int lane = threadIdx.x;
  const int s = lane & 15;
  const int ch = blockIdx.x * 2 + (lane >> 4);
  const float A = -expf(A_log[ch * 16 + s]);
  const float Dch = Dv[ch];
  float h = 0.f;
  for (int l = 0; l < L; ++l) {
    float dtv = dt[(size_t)ch * L + l];
    float uv  = u[(size_t)ch * L + l];
    float a   = expf(dtv * A);
    float bu  = dtv * Bm[(size_t)s * L + l] * uv;
    h = a * h + bu;
    float p = h * Cm[(size_t)s * L + l];
#pragma unroll
    for (int m = 8; m; m >>= 1) p += __shfl_xor(p, m, 32);
    if (s == 0) {
      float zv = z[(size_t)ch * L + l];
      float yv = p + uv * Dch;
      y[(size_t)ch * L + l] = yv * (zv / (1.f + expf(-zv)));
    }
  }
}

// ---------------------------------------------------------------------------
// Host orchestration.
// ---------------------------------------------------------------------------
extern "C" void kernel_launch(void* const* d_in, const int* in_sizes, int n_in,
                              void* d_out, int out_size, void* d_ws, size_t ws_size,
                              hipStream_t stream) {
  const int DIM[4] = {48, 96, 192, 384};
  const int SP[4]  = {32, 16, 8, 4};
  const int LW[4]  = {5, 4, 3, 2};
  const size_t OUT_OFF[4] = {0, 1572864, 1966080, 2064384};

  // --- parameter pointers, JAX pytree flatten order (dict keys sorted) ---
  struct Rmb { const float *A_log,*Dv,*conv_b,*conv_w,*dt_b,*dt_w,*in_proj,
                           *ln_b,*ln_g,*out_proj,*x_proj; };
  struct Stg { const float *ds_b,*ds_w,*p1b,*p1w,*p2b,*p2w,*p3b,*p3w,*p4b,*p4w,
                           *fc1b,*fc1w,*fc2b,*fc2w; Rmb r[2]; };
  const float* x1 = (const float*)d_in[0];
  Stg P[4];
  int k = 1;
  auto nx = [&]() { return (const float*)d_in[k++]; };
  for (int i = 0; i < 4; ++i) {
    P[i].ds_b = nx(); P[i].ds_w = nx();
    P[i].p1b = nx(); P[i].p1w = nx(); P[i].p2b = nx(); P[i].p2w = nx();
    P[i].p3b = nx(); P[i].p3w = nx(); P[i].p4b = nx(); P[i].p4w = nx();
    P[i].fc1b = nx(); P[i].fc1w = nx(); P[i].fc2b = nx(); P[i].fc2w = nx();
    for (int j = 0; j < 2; ++j) {
      Rmb& r = P[i].r[j];
      r.A_log = nx(); r.Dv = nx(); r.conv_b = nx(); r.conv_w = nx();
      r.dt_b = nx(); r.dt_w = nx(); r.in_proj = nx(); r.ln_b = nx();
      r.ln_g = nx(); r.out_proj = nx(); r.x_proj = nx();
    }
  }

  // --- workspace carve-up (floats) ---
  float* w0 = (float*)d_ws;
  size_t off = 0;
  auto alloc = [&](size_t n) { float* p = w0 + off; off += n; return p; };
  float* XA  = alloc(1600000);
  float* XB  = alloc(1600000);
  float* T1  = alloc(1600000);
  float* T2  = alloc(1600000);
  float* T3  = alloc(1600000);
  float* LNb = alloc(1600000);
  float* XZ  = alloc(6300000);   // (2*di, L)
  float* U   = alloc(3200000);   // (di, L)
  float* XDB = alloc(2200000);   // (dtr+32, L), padded to 64 rows
  float* DT  = alloc(3200000);   // (di, L)
  float* Y   = alloc(3200000);   // (di, L)
  float* H1  = alloc(3200000);   // (2d, L)
  float* MEANB = alloc(1024);
  float* RSTDB = alloc(1024);
  (void)ws_size; (void)in_sizes; (void)n_in; (void)out_size;

  auto gemm = [&](const float* in, const float* wgt, const float* bias,
                  float* out, int cin, int cout, int ks, int st, int pd,
                  int si, int so, int lw) {
    dim3 grid(cdiv_h(so * so * so, 128), cdiv_h(cout, 16));
    k_conv_wmma<<<grid, 128, 0, stream>>>(in, wgt, bias, out, cin, cout, ks,
                                          st, pd, si, si, si, so, so, so, lw);
  };
  auto inorm = [&](const float* xin, float* xout, int C, int L, int relu) {
    k_inorm_stats<<<C, 256, 0, stream>>>(xin, MEANB, RSTDB, L);
    k_inorm_apply<<<dim3(cdiv_h(L, 256), C), 256, 0, stream>>>(
        xin, MEANB, RSTDB, xout, L, relu);
  };
  auto add = [&](const float* a, const float* b, float* o, int n) {
    k_add<<<cdiv_h(n, 256), 256, 0, stream>>>(a, b, o, n);
  };

  float* cur = XA;
  // Stage 0 stem: (1,64^3) -> (48, 32^3), channel-major.
  k_stem<<<6144, 256, 0, stream>>>(x1, P[0].ds_w, P[0].ds_b, cur);

  for (int i = 0; i < 4; ++i) {
    const int d = DIM[i], s = SP[i], L = s * s * s, lw = LW[i];
    const int di = 2 * d, dtr = (d + 15) / 16;
    float* nxt = (cur == XA) ? XB : XA;

    if (i > 0) {  // downsample: conv2^3 stride2 of inorm(x)
      const int dp = DIM[i - 1], sp_ = SP[i - 1], Lp = sp_ * sp_ * sp_;
      inorm(cur, T1, dp, Lp, 0);
      gemm(T1, P[i].ds_w, P[i].ds_b, nxt, dp, d, 2, 2, 0, sp_, s, lw);
      cur = nxt; nxt = (cur == XA) ? XB : XA;
    }

    // ---- HGDC block ----
    gemm(cur, P[i].p1w, P[i].p1b, T1, d, d, 3, 1, 1, s, s, lw);
    inorm(T1, T1, d, L, 1);
    gemm(T1, P[i].p2w, P[i].p2b, T2, d, d, 3, 1, 1, s, s, lw);
    inorm(T2, T2, d, L, 1);
    gemm(cur, P[i].p3w, P[i].p3b, T3, d, d, 1, 1, 0, s, s, lw);
    inorm(T3, T3, d, L, 1);
    add(T2, T3, T1, d * L);
    gemm(T1, P[i].p4w, P[i].p4b, T2, d, d, 1, 1, 0, s, s, lw);
    inorm(T2, T2, d, L, 1);
    add(T2, cur, nxt, d * L);
    cur = nxt; nxt = (cur == XA) ? XB : XA;

    // ---- 2 Mamba residual blocks ----
    for (int j = 0; j < 2; ++j) {
      const Rmb& r = P[i].r[j];
      k_lnorm<<<cdiv_h(L, 4), 128, 0, stream>>>(cur, r.ln_g, r.ln_b, LNb, d, L);
      gemm(LNb, r.in_proj, nullptr, XZ, d, 2 * di, 1, 1, 0, s, s, lw);
      // xc = XZ rows [0, di), z = XZ rows [di, 2di)
      k_dwconv<<<cdiv_h(di * L, 256), 256, 0, stream>>>(XZ, r.conv_w, r.conv_b,
                                                        U, di, L);
      gemm(U, r.x_proj, nullptr, XDB, di, dtr + 32, 1, 1, 0, s, s, lw);
      gemm(XDB, r.dt_w, r.dt_b, DT, dtr, di, 1, 1, 0, s, s, lw);
      k_softplus<<<cdiv_h(di * L, 256), 256, 0, stream>>>(DT, di * L);
      k_scan<<<di / 2, 32, 0, stream>>>(DT, XDB + (size_t)dtr * L,
                                        XDB + (size_t)(dtr + 16) * L, U,
                                        r.A_log, r.Dv, XZ + (size_t)di * L, Y, L);
      gemm(Y, r.out_proj, nullptr, T1, di, d, 1, 1, 0, s, s, lw);
      add(T1, cur, nxt, d * L);
      cur = nxt; nxt = (cur == XA) ? XB : XA;
    }

    // ---- MLP head -> stage output ----
    inorm(cur, T1, d, L, 0);
    gemm(T1, P[i].fc1w, P[i].fc1b, H1, d, 2 * d, 1, 1, 0, s, s, lw);
    k_gelu<<<cdiv_h(2 * d * L, 256), 256, 0, stream>>>(H1, 2 * d * L);
    gemm(H1, P[i].fc2w, P[i].fc2b, (float*)d_out + OUT_OFF[i], 2 * d, d,
         1, 1, 0, s, s, lw);
  }
}
